// MaskGIT_6743098655164
// MI455X (gfx1250) — matrix operand order repaired and checked by reading, not compiled, not run on hardware
//
#include <hip/hip_runtime.h>
#include <cmath>
#include <cstddef>

#define MASK_TOK  (-1)
#define BATCH     4
#define SEQ       1024
#define NROWS     (BATCH*SEQ)     /* 4096 */
#define HDIM      512
#define FFDIM     2048
#define KCB       1024
#define NHEADS    8
#define DHEAD     64
#define NLAYERS   2
#define NSTEPS    4
#define LN_EPS    1e-3f

typedef __attribute__((ext_vector_type(16))) __bf16 v16bf;
typedef __attribute__((ext_vector_type(8)))  float  v8f;

union Frag  { v16bf v; unsigned u[8]; };
union FragF { v8f   v; float    f[8]; };
union Pack8 { uint4 q; unsigned short s[8]; };

__device__ __forceinline__ unsigned short f2bf(float f){
  unsigned u = __float_as_uint(f);
  u += 0x7FFFu + ((u >> 16) & 1u);        // round-to-nearest-even
  return (unsigned short)(u >> 16);
}
__device__ __forceinline__ float gelu_tanh(float x){
  float x3 = x*x*x;
  return 0.5f*x*(1.0f + tanhf(0.7978845608028654f*(x + 0.044715f*x3)));
}
__device__ __forceinline__ v8f wmma_bf16(v16bf a, v16bf b, v8f c){
  return __builtin_amdgcn_wmma_f32_16x16x32_bf16(false, a, false, b, (short)0, c, false, false);
}
// low 32 bits of a generic pointer to __shared__ == addrspace(3) byte offset (LDS_BASE-relative)
__device__ __forceinline__ unsigned lds_off32(const void* p){
  return (unsigned)(unsigned long long)(size_t)p;
}

// ---------------- elementwise / setup kernels ----------------

__global__ void k_f32_to_bf16(const float* __restrict__ src, unsigned short* __restrict__ dst, int n){
  int i = blockIdx.x*blockDim.x + threadIdx.x;
  if (i < n) dst[i] = f2bf(src[i]);
}

__global__ void k_zero_f32(float* dst, int n){
  int i = blockIdx.x*blockDim.x + threadIdx.x;
  if (i < n) dst[i] = 0.0f;
}

__global__ void k_init_small(int* iu, float* thr){
  if (threadIdx.x < BATCH){ iu[threadIdx.x] = 0; thr[threadIdx.x] = 0.0f; }
}

__global__ void k_init_tokens(const int* __restrict__ in, int* __restrict__ tok, int* __restrict__ iu){
  int i = blockIdx.x*blockDim.x + threadIdx.x;
  if (i < NROWS){
    int t = in[i];
    tok[i] = t;
    if (t == MASK_TOK) atomicAdd(&iu[i >> 10], 1);
  }
}

__global__ void k_embed(const int* __restrict__ tok, const float* __restrict__ cb,
                        const float* __restrict__ pos, float* __restrict__ out){
  int row = blockIdx.x;
  int t = tok[row];
  int s = row & (SEQ-1);
  for (int c = threadIdx.x; c < HDIM; c += blockDim.x){
    float e = (t == MASK_TOK) ? 0.0f : cb[(size_t)t*HDIM + c];
    out[(size_t)row*HDIM + c] = e + pos[(size_t)s*HDIM + c];
  }
}

// y = LN(a (+ res)) * g + b ; optional f32 and bf16 outputs. One 128-thread block per row.
__global__ __launch_bounds__(128) void k_layernorm(const float* __restrict__ a, const float* __restrict__ res,
    const float* __restrict__ g, const float* __restrict__ beta,
    float* __restrict__ y32, unsigned short* __restrict__ ybf){
  __shared__ float sh[128];
  int row = blockIdx.x, tid = threadIdx.x;
  float v[4]; float s = 0.0f;
  #pragma unroll
  for (int t=0;t<4;t++){
    int c = tid + t*128;
    float x = a[(size_t)row*HDIM + c];
    if (res) x += res[(size_t)row*HDIM + c];
    v[t]=x; s+=x;
  }
  sh[tid]=s; __syncthreads();
  for (int off=64; off>0; off>>=1){ if (tid<off) sh[tid]+=sh[tid+off]; __syncthreads(); }
  float mean = sh[0]*(1.0f/HDIM); __syncthreads();
  float s2=0.0f;
  #pragma unroll
  for (int t=0;t<4;t++){ float d=v[t]-mean; s2+=d*d; }
  sh[tid]=s2; __syncthreads();
  for (int off=64; off>0; off>>=1){ if (tid<off) sh[tid]+=sh[tid+off]; __syncthreads(); }
  float inv = rsqrtf(sh[0]*(1.0f/HDIM) + LN_EPS); __syncthreads();
  #pragma unroll
  for (int t=0;t<4;t++){
    int c = tid + t*128;
    float y = (v[t]-mean)*inv*g[c] + beta[c];
    if (y32) y32[(size_t)row*HDIM + c] = y;
    if (ybf) ybf[(size_t)row*HDIM + c] = f2bf(y);
  }
}

// ---------------- WMMA GEMM: C[M,N] = A[M,K](bf16) * W[K,N](bf16) + bias ----------------
// block tile 64x128, 8 waves (2x4), each wave 32x32 = 2x2 wmma accumulators.
#define BM 64
#define BN 128
#define BK 32

__global__ __launch_bounds__(256) void k_gemm(const unsigned short* __restrict__ A,
    const unsigned short* __restrict__ W, const float* __restrict__ bias,
    float* __restrict__ outF, unsigned short* __restrict__ outB,
    int M, int K, int N, int act){
  __shared__ unsigned short As[BM][BK];   // row-major (k pairs contiguous)
  __shared__ unsigned short Bs[BN][BK];   // transposed: [n][k]
  int tid = threadIdx.x;
  int wid = tid >> 5, lane = tid & 31;
  int lm = lane & 15, lh = lane >> 4;
  int m0 = blockIdx.y * BM;
  int n0 = blockIdx.x * BN;
  int wm = (wid & 1) * 32;
  int wn = (wid >> 1) * 32;

  FragF acc[2][2];
  #pragma unroll
  for (int i=0;i<2;i++)
    #pragma unroll
    for (int j=0;j<2;j++)
      #pragma unroll
      for (int v=0;v<8;v++) acc[i][j].f[v] = 0.0f;

  // per-thread staging coordinates
  int ar  = tid >> 2;            // A: row 0..63
  int acg = (tid & 3) * 8;       // A: col group (8 bf16 = b128)
  int kp  = tid >> 4;            // B: k-pair 0..15 -> rows 2kp, 2kp+1
  int ng  = (tid & 15) * 8;      // B: 8 columns

  for (int kk = 0; kk < K; kk += BK){
    { // A tile 64x32: async copy global -> LDS, one b128 per thread (ASYNCcnt path)
      const unsigned short* gsrc = A + (size_t)(m0 + ar)*K + kk + acg;
      unsigned dst = lds_off32(&As[ar][acg]);
      asm volatile("global_load_async_to_lds_b128 %0, %1, off"
                   :: "v"(dst), "v"((unsigned long long)(size_t)gsrc) : "memory");
      if (kk + BK < K) __builtin_prefetch(gsrc + BK, 0, 1);
    }
    { // B tile 32x128 transposed into Bs[n][k]: pack (k,k+1) pairs -> dword LDS stores
      Pack8 r0, r1;
      const unsigned short* src0 = W + (size_t)(kk + 2*kp    )*N + n0 + ng;
      const unsigned short* src1 = W + (size_t)(kk + 2*kp + 1)*N + n0 + ng;
      r0.q = *(const uint4*)src0;
      r1.q = *(const uint4*)src1;
      #pragma unroll
      for (int t=0;t<8;t++){
        unsigned packed = (unsigned)r0.s[t] | ((unsigned)r1.s[t] << 16);
        *(unsigned*)&Bs[ng + t][2*kp] = packed;
      }
      if (kk + BK < K) __builtin_prefetch(src0 + (size_t)BK*N, 0, 1);
    }
    asm volatile("s_wait_asynccnt 0x0" ::: "memory");
    __syncthreads();

    Frag a[2], b[2];
    #pragma unroll
    for (int i=0;i<2;i++){
      int mb = wm + i*16 + lm;
      #pragma unroll
      for (int v=0;v<8;v++){
        int k0 = ((v & 4) ? 16 : 0) + lh*8 + (v & 3)*2;   // 16-bit A layout
        a[i].u[v] = *(const unsigned*)&As[mb][k0];
      }
    }
    #pragma unroll
    for (int j=0;j<2;j++){
      int nb = wn + j*16 + lm;
      #pragma unroll
      for (int v=0;v<8;v++){
        int k0 = ((v & 4) ? 16 : 0) + lh*8 + (v & 3)*2;   // 16-bit B layout
        b[j].u[v] = *(const unsigned*)&Bs[nb][k0];
      }
    }
    #pragma unroll
    for (int i=0;i<2;i++)
      #pragma unroll
      for (int j=0;j<2;j++)
        acc[i][j].v = wmma_bf16(a[i].v, b[j].v, acc[i][j].v);
    __syncthreads();
  }

  #pragma unroll
  for (int i=0;i<2;i++)
    #pragma unroll
    for (int j=0;j<2;j++)
      #pragma unroll
      for (int v=0;v<8;v++){
        int row = m0 + wm + i*16 + v + 8*lh;   // C layout: M = v + 8*half
        int col = n0 + wn + j*16 + lm;         // N = lane%16
        float r = acc[i][j].f[v] + bias[col];
        if (act == 1) r = gelu_tanh(r);
        if (outF) outF[(size_t)row*N + col] = r;
        if (outB) outB[(size_t)row*N + col] = f2bf(r);
      }
}

// ---------------- flash attention: per-wave 16 queries, DH=64, keys in chunks of 32 ----
__global__ __launch_bounds__(128) void k_attn(const unsigned short* __restrict__ Q,
    const unsigned short* __restrict__ Kb, const unsigned short* __restrict__ Vb,
    unsigned short* __restrict__ O){
  __shared__ unsigned short P[4][16][32];    // per-wave probs (A re-layout staging)
  __shared__ unsigned short Vt[4][64][32];   // per-wave V^T chunk [d][k]
  int w = threadIdx.x >> 5, lane = threadIdx.x & 31;
  int lm = lane & 15, lh = lane >> 4;
  int blk = blockIdx.x;                      // b*128 + h*16 + qc
  int qc = blk & 15;
  int h  = (blk >> 4) & 7;
  int b  = blk >> 7;
  int q0 = qc*64 + w*16;
  size_t rowbase = (size_t)b * SEQ;
  int coff = h * DHEAD;

  Frag qf[2];
  #pragma unroll
  for (int kc=0;kc<2;kc++)
    #pragma unroll
    for (int v=0;v<8;v++){
      int k0 = ((v & 4) ? 16 : 0) + lh*8 + (v & 3)*2 + kc*32;
      qf[kc].u[v] = *(const unsigned*)(Q + (rowbase + q0 + lm)*HDIM + coff + k0);
    }

  float mrun[8], lrun[8];
  FragF o[4];
  #pragma unroll
  for (int v=0;v<8;v++){ mrun[v] = -3.0e38f; lrun[v] = 0.0f; }
  #pragma unroll
  for (int t=0;t<4;t++)
    #pragma unroll
    for (int v=0;v<8;v++) o[t].f[v] = 0.0f;

  for (int jb = 0; jb < SEQ; jb += 32){
    // stage V^T for this chunk
    for (int r=0;r<32;r++){
      unsigned uv = *(const unsigned*)(Vb + (rowbase + jb + r)*HDIM + coff + 2*lane);
      Vt[w][2*lane  ][r] = (unsigned short)(uv & 0xFFFFu);
      Vt[w][2*lane+1][r] = (unsigned short)(uv >> 16);
    }
    // scores: S[16x32] = Q(16x64) . K_chunk^T   (2 n-tiles x 2 k-slabs)
    FragF s0, s1;
    #pragma unroll
    for (int v=0;v<8;v++){ s0.f[v]=0.0f; s1.f[v]=0.0f; }
    #pragma unroll
    for (int kc=0;kc<2;kc++){
      Frag bk0, bk1;
      #pragma unroll
      for (int v=0;v<8;v++){
        int k0 = ((v & 4) ? 16 : 0) + lh*8 + (v & 3)*2 + kc*32;
        bk0.u[v] = *(const unsigned*)(Kb + (rowbase + jb +      lm)*HDIM + coff + k0);
        bk1.u[v] = *(const unsigned*)(Kb + (rowbase + jb + 16 + lm)*HDIM + coff + k0);
      }
      s0.v = wmma_bf16(qf[kc].v, bk0.v, s0.v);
      s1.v = wmma_bf16(qf[kc].v, bk1.v, s1.v);
    }
    // online softmax (row = v + 8*half, spread across a 16-lane half)
    float alpha[8];
    #pragma unroll
    for (int v=0;v<8;v++){
      float a0 = s0.f[v]*0.125f, a1 = s1.f[v]*0.125f;
      float mx = fmaxf(a0, a1);
      #pragma unroll
      for (int off=1; off<16; off<<=1) mx = fmaxf(mx, __shfl_xor(mx, off));
      float mn = fmaxf(mrun[v], mx);
      alpha[v] = expf(mrun[v] - mn);
      float p0 = expf(a0 - mn), p1 = expf(a1 - mn);
      float ps = p0 + p1;
      #pragma unroll
      for (int off=1; off<16; off<<=1) ps += __shfl_xor(ps, off);
      lrun[v] = lrun[v]*alpha[v] + ps;
      mrun[v] = mn;
      int prow = v + 8*lh;
      P[w][prow][lm]      = f2bf(p0);
      P[w][prow][lm + 16] = f2bf(p1);
    }
    #pragma unroll
    for (int t=0;t<4;t++)
      #pragma unroll
      for (int v=0;v<8;v++) o[t].f[v] *= alpha[v];
    // P as A-fragment
    Frag pf;
    #pragma unroll
    for (int v=0;v<8;v++){
      int k0 = ((v & 4) ? 16 : 0) + lh*8 + (v & 3)*2;
      pf.u[v] = *(const unsigned*)&P[w][lm][k0];
    }
    // O += P(16x32) . V(32x64) : 4 d-tiles
    #pragma unroll
    for (int t=0;t<4;t++){
      Frag vf;
      #pragma unroll
      for (int v=0;v<8;v++){
        int k0 = ((v & 4) ? 16 : 0) + lh*8 + (v & 3)*2;
        vf.u[v] = *(const unsigned*)&Vt[w][t*16 + lm][k0];
      }
      o[t].v = wmma_bf16(pf.v, vf.v, o[t].v);
    }
  }
  // normalize and emit bf16
  #pragma unroll
  for (int t=0;t<4;t++)
    #pragma unroll
    for (int v=0;v<8;v++){
      size_t row = rowbase + q0 + v + 8*lh;
      int col = coff + t*16 + lm;
      O[row*HDIM + col] = f2bf(o[t].f[v] / lrun[v]);
    }
}

// ---------------- MaskGIT decode step kernels ----------------

// per-token: argmax (first-max tie-break), confidence = -log(sum exp(l - max)) if masked else +inf
__global__ __launch_bounds__(256) void k_conf(const float* __restrict__ logits,
    const int* __restrict__ tok, float* __restrict__ conf, int* __restrict__ samp){
  __shared__ float sm[256];
  __shared__ int   sa[256];
  int row = blockIdx.x, tid = threadIdx.x;
  const float* lr = logits + (size_t)row*KCB;
  float m = -3.0e38f; int arg = 0;
  for (int c = tid; c < KCB; c += 256){
    float v = lr[c];
    if (v > m){ m = v; arg = c; }
  }
  sm[tid]=m; sa[tid]=arg; __syncthreads();
  for (int off=128; off>0; off>>=1){
    if (tid < off){
      float v2 = sm[tid+off]; int a2 = sa[tid+off];
      if (v2 > sm[tid] || (v2 == sm[tid] && a2 < sa[tid])){ sm[tid]=v2; sa[tid]=a2; }
    }
    __syncthreads();
  }
  float gm = sm[0]; int garg = sa[0]; __syncthreads();
  float s = 0.0f;
  for (int c = tid; c < KCB; c += 256) s += expf(lr[c] - gm);
  sm[tid]=s; __syncthreads();
  for (int off=128; off>0; off>>=1){ if (tid<off) sm[tid]+=sm[tid+off]; __syncthreads(); }
  if (tid == 0){
    bool unk = (tok[row] == MASK_TOK);
    conf[row] = unk ? (-logf(sm[0])) : __builtin_inff();
    samp[row] = garg;
  }
}

// per-batch bitonic sort of 1024 confidences; thr = sorted[mask_len]
__global__ __launch_bounds__(512) void k_thresh(const float* __restrict__ conf,
    const int* __restrict__ iu, float* __restrict__ thr, int step, float cosr, int is_last){
  __shared__ float arr[SEQ];
  int b = blockIdx.x;
  for (int i = threadIdx.x; i < SEQ; i += 512) arr[i] = conf[b*SEQ + i];
  __syncthreads();
  for (int k = 2; k <= SEQ; k <<= 1)
    for (int j = k >> 1; j > 0; j >>= 1){
      for (int i = threadIdx.x; i < SEQ; i += 512){
        int l = i ^ j;
        if (l > i){
          bool up = ((i & k) == 0);
          float a = arr[i], c = arr[l];
          if (up ? (a > c) : (a < c)){ arr[i] = c; arr[l] = a; }
        }
      }
      __syncthreads();
    }
  if (threadIdx.x == 0){
    int ml = 0;
    if (!is_last){
      int f = (int)floorf((float)iu[b] * cosr);
      ml = (step + 1 > f) ? (step + 1) : f;
    }
    if (ml < 0) ml = 0;
    if (ml > SEQ-1) ml = SEQ-1;
    thr[b] = arr[ml];
  }
}

__global__ void k_write_logits(const float* __restrict__ logits, const int* __restrict__ tok,
    const float* __restrict__ conf, const float* __restrict__ thr, float* __restrict__ out){
  int idx = blockIdx.x*blockDim.x + threadIdx.x;
  if (idx < NROWS*KCB){
    int row = idx >> 10;
    if (tok[row] == MASK_TOK && conf[row] >= thr[row >> 10]) out[idx] = logits[idx];
  }
}

__global__ void k_update_tokens(int* __restrict__ tok, const float* __restrict__ conf,
    const float* __restrict__ thr, const int* __restrict__ samp){
  int i = blockIdx.x*blockDim.x + threadIdx.x;
  if (i < NROWS && tok[i] == MASK_TOK && conf[i] >= thr[i >> 10]) tok[i] = samp[i];
}

__global__ void k_tokens_out(const int* __restrict__ tok, float* __restrict__ out){
  int i = blockIdx.x*blockDim.x + threadIdx.x;
  if (i < NROWS) out[i] = (float)tok[i];
}

// ---------------- host orchestration ----------------

extern "C" void kernel_launch(void* const* d_in, const int* in_sizes, int n_in,
                              void* d_out, int out_size, void* d_ws, size_t ws_size,
                              hipStream_t stream){
  (void)in_sizes; (void)n_in; (void)out_size; (void)ws_size;
  const int*   tok_in  = (const int*)d_in[0];
  const float* cb      = (const float*)d_in[1];
  const float* pos     = (const float*)d_in[2];
  const float* ln_in_g = (const float*)d_in[3];
  const float* ln_in_b = (const float*)d_in[4];

  const float *wf[2][6], *bvec[2][6], *lng[2][2], *lnb[2][2];
  for (int l = 0; l < NLAYERS; ++l){
    int base = 5 + l*16;
    wf[l][0]=(const float*)d_in[base+0];  bvec[l][0]=(const float*)d_in[base+1];   // wq,bq
    wf[l][1]=(const float*)d_in[base+2];  bvec[l][1]=(const float*)d_in[base+3];   // wk,bk
    wf[l][2]=(const float*)d_in[base+4];  bvec[l][2]=(const float*)d_in[base+5];   // wv,bv
    wf[l][3]=(const float*)d_in[base+6];  bvec[l][3]=(const float*)d_in[base+7];   // wo,bo
    lng[l][0]=(const float*)d_in[base+8]; lnb[l][0]=(const float*)d_in[base+9];
    wf[l][4]=(const float*)d_in[base+10]; bvec[l][4]=(const float*)d_in[base+11];  // w1,b1
    wf[l][5]=(const float*)d_in[base+12]; bvec[l][5]=(const float*)d_in[base+13];  // w2,b2
    lng[l][1]=(const float*)d_in[base+14]; lnb[l][1]=(const float*)d_in[base+15];
  }
  const float* wd   = (const float*)d_in[37];
  const float* bd   = (const float*)d_in[38];
  const float* lnfg = (const float*)d_in[39];
  const float* lnfb = (const float*)d_in[40];
  const float* wout = (const float*)d_in[41];
  const float* bout = (const float*)d_in[42];

  char* p = (char*)d_ws;
  auto alloc = [&](size_t bytes)->char*{
    char* r = p; p += (bytes + 255) & ~(size_t)255; return r;
  };
  int wsz[6] = {HDIM*HDIM, HDIM*HDIM, HDIM*HDIM, HDIM*HDIM, HDIM*FFDIM, FFDIM*HDIM};
  unsigned short* wB[2][6];
  for (int l=0;l<NLAYERS;l++) for (int j=0;j<6;j++) wB[l][j]=(unsigned short*)alloc((size_t)wsz[j]*2);
  unsigned short* wdB   = (unsigned short*)alloc((size_t)HDIM*HDIM*2);
  unsigned short* woutB = (unsigned short*)alloc((size_t)HDIM*KCB*2);
  float*          x     = (float*)alloc((size_t)NROWS*HDIM*4);
  unsigned short* xb    = (unsigned short*)alloc((size_t)NROWS*HDIM*2);
  unsigned short* qb    = (unsigned short*)alloc((size_t)NROWS*HDIM*2);
  unsigned short* kb    = (unsigned short*)alloc((size_t)NROWS*HDIM*2);
  unsigned short* vb    = (unsigned short*)alloc((size_t)NROWS*HDIM*2);
  unsigned short* ob    = (unsigned short*)alloc((size_t)NROWS*HDIM*2);
  unsigned short* hb    = (unsigned short*)alloc((size_t)NROWS*FFDIM*2);
  float*          ftmp  = (float*)alloc((size_t)NROWS*HDIM*4);
  float*          logits= (float*)alloc((size_t)NROWS*KCB*4);
  int*            tok   = (int*)alloc((size_t)NROWS*4);
  int*            samp  = (int*)alloc((size_t)NROWS*4);
  float*          conf  = (float*)alloc((size_t)NROWS*4);
  int*            iu    = (int*)alloc(64);
  float*          thr   = (float*)alloc(64);

  float* out_tok    = (float*)d_out;
  float* out_logits = out_tok + NROWS;

  auto cvt = [&](const float* s, unsigned short* d, int n){
    k_f32_to_bf16<<<(n + 255)/256, 256, 0, stream>>>(s, d, n);
  };
  for (int l=0;l<NLAYERS;l++) for (int j=0;j<6;j++) cvt(wf[l][j], wB[l][j], wsz[j]);
  cvt(wd, wdB, HDIM*HDIM);
  cvt(wout, woutB, HDIM*KCB);

  k_zero_f32<<<(NROWS*KCB + 255)/256, 256, 0, stream>>>(out_logits, NROWS*KCB);
  k_init_small<<<1, 32, 0, stream>>>(iu, thr);
  k_init_tokens<<<(NROWS + 255)/256, 256, 0, stream>>>(tok_in, tok, iu);

  auto gemm = [&](const unsigned short* A, const unsigned short* W, const float* bias,
                  float* oF, unsigned short* oB, int Kd, int Nd, int act){
    dim3 g(Nd/BN, NROWS/BM);
    k_gemm<<<g, 256, 0, stream>>>(A, W, bias, oF, oB, NROWS, Kd, Nd, act);
  };

  for (int step = 0; step < NSTEPS; ++step){
    k_embed<<<NROWS, 128, 0, stream>>>(tok, cb, pos, ftmp);
    k_layernorm<<<NROWS, 128, 0, stream>>>(ftmp, nullptr, ln_in_g, ln_in_b, x, xb);
    for (int l = 0; l < NLAYERS; ++l){
      gemm(xb, wB[l][0], bvec[l][0], nullptr, qb, HDIM, HDIM, 0);
      gemm(xb, wB[l][1], bvec[l][1], nullptr, kb, HDIM, HDIM, 0);
      gemm(xb, wB[l][2], bvec[l][2], nullptr, vb, HDIM, HDIM, 0);
      k_attn<<<BATCH*NHEADS*(SEQ/64), 128, 0, stream>>>(qb, kb, vb, ob);
      gemm(ob, wB[l][3], bvec[l][3], ftmp, nullptr, HDIM, HDIM, 0);
      k_layernorm<<<NROWS, 128, 0, stream>>>(x, ftmp, lng[l][0], lnb[l][0], x, xb);
      gemm(xb, wB[l][4], bvec[l][4], nullptr, hb, HDIM, FFDIM, 1);   // gelu
      gemm(hb, wB[l][5], bvec[l][5], ftmp, nullptr, FFDIM, HDIM, 0);
      k_layernorm<<<NROWS, 128, 0, stream>>>(x, ftmp, lng[l][1], lnb[l][1], x, xb);
    }
    gemm(xb, wdB, bd, ftmp, nullptr, HDIM, HDIM, 1);                  // gelu
    k_layernorm<<<NROWS, 128, 0, stream>>>(ftmp, nullptr, lnfg, lnfb, nullptr, ob);
    gemm(ob, woutB, bout, logits, nullptr, HDIM, KCB, 0);

    k_conf<<<NROWS, 256, 0, stream>>>(logits, tok, conf, samp);
    float ratio = (float)(step + 1) / (float)NSTEPS;
    float cosr  = cosf(1.57079632679489662f * ratio);
    k_thresh<<<BATCH, 512, 0, stream>>>(conf, iu, thr, step, cosr, (step == NSTEPS-1) ? 1 : 0);
    k_write_logits<<<(NROWS*KCB + 255)/256, 256, 0, stream>>>(logits, tok, conf, thr, out_logits);
    k_update_tokens<<<(NROWS + 255)/256, 256, 0, stream>>>(tok, conf, thr, samp);
  }
  k_tokens_out<<<(NROWS + 255)/256, 256, 0, stream>>>(tok, out_tok);
}
